// GroupedESN_72782515799000
// MI455X (gfx1250) — matrix operand order repaired and checked by reading, not compiled, not run on hardware
//
#include <hip/hip_runtime.h>
#include <hip/hip_bf16.h>
#include <cstdint>
#include <cstddef>

// ---- problem constants (from reference) ----
#define E_ 8
#define B_ 16
#define T_ 512
#define R_ 512
#define D_ 8
#define HPAD 520   // padded row stride (bf16 elems): 520*2B = 1040B = 260 dwords -> +4 banks/row

typedef __bf16 v8bf  __attribute__((ext_vector_type(8)));
typedef __bf16 v16bf __attribute__((ext_vector_type(16)));
typedef float  v2f   __attribute__((ext_vector_type(2)));
typedef float  v8f   __attribute__((ext_vector_type(8)));

__device__ __forceinline__ unsigned short f32_to_bf16_rne(float f) {
  unsigned int u = __float_as_uint(f);
  u += 0x7FFFu + ((u >> 16) & 1u);   // round to nearest even
  return (unsigned short)(u >> 16);
}

__device__ __forceinline__ float fast_tanh(float x) {
#if __has_builtin(__builtin_amdgcn_tanhf)
  return __builtin_amdgcn_tanhf(x);   // gfx1250 hardware v_tanh_f32
#else
  return tanhf(x);
#endif
}

// ---- prep: convert W_res fp32 -> bf16 into workspace ----
__global__ void wres_to_bf16_kernel(const float* __restrict__ src,
                                    unsigned short* __restrict__ dst,
                                    int n) {
  int i = (blockIdx.x * blockDim.x + threadIdx.x) * 8;
  if (i + 7 >= n) return;
  float4 a = *(const float4*)(src + i);
  float4 b = *(const float4*)(src + i + 4);
  union { unsigned short us[8]; uint4 v; } p;
  p.us[0] = f32_to_bf16_rne(a.x); p.us[1] = f32_to_bf16_rne(a.y);
  p.us[2] = f32_to_bf16_rne(a.z); p.us[3] = f32_to_bf16_rne(a.w);
  p.us[4] = f32_to_bf16_rne(b.x); p.us[5] = f32_to_bf16_rne(b.y);
  p.us[6] = f32_to_bf16_rne(b.z); p.us[7] = f32_to_bf16_rne(b.w);
  *(uint4*)(dst + i) = p.v;
}

// ---- main: one workgroup (8 waves) per expert; sequential scan over T ----
// W_res residency: rows [0,256) in LDS, rows [256,512) in persistent VGPR
// fragments (2 n-tiles x 16 k-tiles x 8 VGPRs per wave). Hot loop touches no
// global memory except the 512B x-prefetch.
__global__ __launch_bounds__(256, 1)
void esn_scan_kernel(const float* __restrict__ x,              // [B][T][D] f32
                     const float* __restrict__ W_in,           // [E][R][D] f32
                     const unsigned short* __restrict__ Wres,  // [E][R][R] bf16
                     const float* __restrict__ lr,             // [E]
                     float* __restrict__ out)                  // [B][E*R] f32
{
  __shared__ unsigned short Wres_lds[256 * HPAD];  // rows [0,256), padded
  __shared__ unsigned short h_bf[B_ * HPAD];       // bf16 mirror of h
  __shared__ float x_tile[B_][D_];

  const int e    = blockIdx.x;
  const int tid  = threadIdx.x;
  const int lane = tid & 31;
  const int w    = tid >> 5;      // wave 0..7
  const int half = lane >> 4;     // 0: lanes 0-15, 1: lanes 16-31
  const int l15  = lane & 15;

  const float a_lr = lr[e];
  const float om_a = 1.0f - a_lr;

  const unsigned short* Wres_g = Wres + (size_t)e * R_ * R_;

  // ---- stage W_res rows [0,256) into LDS (16B chunks, padded rows) ----
  for (int c = tid; c < 256 * (R_ / 8); c += 256) {
    int row = c >> 6;              // 64 chunks of 8 bf16 per 512-wide row
    int col = (c & 63) * 8;
    *(uint4*)(&Wres_lds[row * HPAD + col]) =
        *(const uint4*)(Wres_g + (size_t)row * R_ + col);
  }
  // ---- h = 0 (bf16 mirror) ----
  for (int c = tid; c < B_ * (R_ / 8); c += 256) {
    int row = c >> 6;
    int col = (c & 63) * 8;
    *(uint4*)(&h_bf[row * HPAD + col]) = make_uint4(0u, 0u, 0u, 0u);
  }
  // ---- x tile for t=0 ----
  if (tid < B_) {
    const float4* xs = (const float4*)(x + (size_t)tid * T_ * D_);
    *(float4*)(&x_tile[tid][0]) = xs[0];
    *(float4*)(&x_tile[tid][4]) = xs[1];
  }

  // ---- per-lane constants: output columns + W_in^T fragments ----
  int rj[4];
  v2f win[4][2];
#pragma unroll
  for (int j = 0; j < 4; ++j) {
    rj[j] = (j * 8 + w) * 16 + l15;          // j<2 -> rows <256 (LDS), j>=2 -> VGPR-resident
    const float* wp = W_in + ((size_t)e * R_ + rj[j]) * D_;
    win[j][0] = *(const v2f*)(wp + half * 2);        // B[k=half*2+{0,1}][n=rj]
    win[j][1] = *(const v2f*)(wp + 4 + half * 2);    // d=4..7 chunk
  }

  // ---- persistent VGPR-resident B fragments for rows [256,512) ----
  // bres[j][kt]: B tile (32x16) for n-tile rj[2+j], K-tile kt.
  v16bf bres[2][16];
#pragma unroll
  for (int j = 0; j < 2; ++j) {
    const unsigned short* bp = Wres_g + (size_t)rj[2 + j] * R_ + half * 16;
#pragma unroll
    for (int kt = 0; kt < 16; ++kt) {
      bres[j][kt] = *(const v16bf*)(bp + kt * 32);
    }
  }

  // h fragments (f32, C/D layout: row M = i + half*8, col N = rj[j])
  v8f hf[4];
#pragma unroll
  for (int j = 0; j < 4; ++j) hf[j] = (v8f){};

  __syncthreads();

  for (int t = 0; t < T_; ++t) {
    // input projection via f32 WMMA: A = x_t (16x4 per call), B = W_in^T
    v2f xa0 = *(const v2f*)(&x_tile[l15][half * 2]);
    v2f xa1 = *(const v2f*)(&x_tile[l15][4 + half * 2]);

    v8f acc[4];
#pragma unroll
    for (int j = 0; j < 4; ++j) {
      v8f c = (v8f){};
      c = __builtin_amdgcn_wmma_f32_16x16x4_f32(false, xa0, false, win[j][0],
                                                (short)0, c, false, false);
      c = __builtin_amdgcn_wmma_f32_16x16x4_f32(false, xa1, false, win[j][1],
                                                (short)0, c, false, false);
      acc[j] = c;
    }

    // reservoir GEMM: acc += h(16x512) * W_res^T ; K tiles of 32, full unroll
#pragma unroll
    for (int kt = 0; kt < 16; ++kt) {
      // A fragment from h mirror in LDS: two 16B chunks per lane
      const unsigned short* ap = &h_bf[l15 * HPAD + kt * 32 + half * 16];
      v8bf alo = *(const v8bf*)(ap);
      v8bf ahi = *(const v8bf*)(ap + 8);
      v16bf af = __builtin_shufflevector(alo, ahi, 0, 1, 2, 3, 4, 5, 6, 7,
                                         8, 9, 10, 11, 12, 13, 14, 15);
      const int kb = kt * 32 + half * 16;  // 16 contiguous K per lane (B layout)

      // LDS-resident tiles (output rows < 256)
      {
        v16bf b0 = *(const v16bf*)(&Wres_lds[rj[0] * HPAD + kb]);
        acc[0] = __builtin_amdgcn_wmma_f32_16x16x32_bf16(
            false, af, false, b0, (short)0, acc[0], false, false);
        v16bf b1 = *(const v16bf*)(&Wres_lds[rj[1] * HPAD + kb]);
        acc[1] = __builtin_amdgcn_wmma_f32_16x16x32_bf16(
            false, af, false, b1, (short)0, acc[1], false, false);
      }
      // VGPR-resident tiles (output rows >= 256): no memory access at all
      acc[2] = __builtin_amdgcn_wmma_f32_16x16x32_bf16(
          false, af, false, bres[0][kt], (short)0, acc[2], false, false);
      acc[3] = __builtin_amdgcn_wmma_f32_16x16x32_bf16(
          false, af, false, bres[1][kt], (short)0, acc[3], false, false);
    }

    __syncthreads();   // all reads of h_bf for step t complete

    // leaky update in f32 registers; publish bf16 mirror for step t+1
#pragma unroll
    for (int j = 0; j < 4; ++j) {
#pragma unroll
      for (int i = 0; i < 8; ++i) {
        float hn = om_a * hf[j][i] + a_lr * fast_tanh(acc[j][i]);
        hf[j][i] = hn;
        int b = i + half * 8;
        h_bf[b * HPAD + rj[j]] = f32_to_bf16_rne(hn);
      }
    }
    // prefetch x_{t+1}
    if (tid < B_ && (t + 1) < T_) {
      const float4* xs = (const float4*)(x + ((size_t)tid * T_ + (t + 1)) * D_);
      *(float4*)(&x_tile[tid][0]) = xs[0];
      *(float4*)(&x_tile[tid][4]) = xs[1];
    }
    __syncthreads();   // h_bf / x_tile visible for step t+1
  }

  // out[b][e*R + r] = h_T[e][b][r]
#pragma unroll
  for (int j = 0; j < 4; ++j) {
#pragma unroll
    for (int i = 0; i < 8; ++i) {
      int b = i + half * 8;
      out[(size_t)b * (E_ * R_) + (size_t)e * R_ + rj[j]] = hf[j][i];
    }
  }
}

extern "C" void kernel_launch(void* const* d_in, const int* in_sizes, int n_in,
                              void* d_out, int out_size, void* d_ws, size_t ws_size,
                              hipStream_t stream) {
  (void)in_sizes; (void)n_in; (void)out_size; (void)ws_size;
  const float* x     = (const float*)d_in[0];   // [16][512][8]
  const float* W_in  = (const float*)d_in[1];   // [8][512][8]
  const float* W_res = (const float*)d_in[2];   // [8][512][512]
  const float* lr    = (const float*)d_in[3];   // [8]
  float* out = (float*)d_out;                   // [16][4096]
  unsigned short* wres_bf = (unsigned short*)d_ws;  // 4 MB bf16 W_res

  const int nW = E_ * R_ * R_;                  // 2,097,152
  wres_to_bf16_kernel<<<nW / (256 * 8), 256, 0, stream>>>(W_res, wres_bf, nW);
  esn_scan_kernel<<<E_, 256, 0, stream>>>(x, W_in, wres_bf, lr, out);
}
